// SSMLayer_57784490001037
// MI455X (gfx1250) — compile-verified, matrix operand order checked by reading, not run on hardware
//
#include <hip/hip_runtime.h>
#include <hip/hip_bf16.h>

// ---------------------------------------------------------------------------
// SSM layer for MI455X (gfx1250, wave32, WMMA bf16 16x16x32).
// Pipeline:
//   prep_x      : x f32 -> bf16
//   prep_w      : exp(log_A)->P1/P1T bf16, bf16 copies of B_w/C_w/gate_w/out_w
//   k_pow  x6   : log-depth matrix powers P_j = A^j (row-major + transposed)
//   k_gemm_Bx   : Bx = x @ B_w^T + b          (WMMA)
//   k_conv      : grouped causal conv (async-to-LDS staging + VALU) -> bf16
//   k_gemm_gate : comb[:,512:] = conv * sigmoid(x @ gate_w^T + b)   (WMMA)
//   k_scan1     : per-chunk local scan (h in LDS, A streamed from L2) (WMMA)
//   k_scan2     : sequential carry across 128 chunks (1 WG)          (WMMA)
//   k_corr      : hs(i,j) += E(i-1) @ (A^{j+1})^T                    (WMMA)
//   k_gemm_C    : comb[:,:512] = hs @ C_w^T + c + x*D                (WMMA)
//   k_gemm_out  : y = comb @ out_w^T + b + x                         (WMMA)
//   k_ln        : LayerNorm(y) -> d_out
// Scratch requirement: ~522 MiB in d_ws (see layout in kernel_launch).
// ---------------------------------------------------------------------------

typedef __bf16 bf16;
typedef __attribute__((ext_vector_type(16))) __bf16 v16bf;
typedef __attribute__((ext_vector_type(8)))  __bf16 v8bf;
typedef __attribute__((ext_vector_type(8)))  float  v8f;

#define DEV static __device__ __forceinline__

constexpr int Bsz  = 8;
constexpr int Lseq = 8192;
constexpr int Dm   = 512;
constexpr int Kc   = 4;
constexpr int Gc   = 8;
constexpr int Dg   = Dm / Gc;           // 64
constexpr int Mtot = Bsz * Lseq;        // 65536
constexpr int CH   = 64;                // scan chunk length
constexpr int NCH  = Lseq / CH;         // 128 chunks
constexpr float LN_EPS = 1e-5f;

DEV v8f vzero() {
  v8f z;
#pragma unroll
  for (int i = 0; i < 8; ++i) z[i] = 0.0f;
  return z;
}

DEV v8f wmma_bf16(v16bf a, v16bf b, v8f c) {
  // D = A(16x32 bf16) * B(32x16 bf16) + C(16x16 f32)
  return __builtin_amdgcn_wmma_f32_16x16x32_bf16(false, a, false, b,
                                                 (short)0, c, false, false);
}

// Async copy: 16 bytes global -> LDS, tracked by ASYNCcnt (gfx1250).
// vdst = LDS byte offset (low 32 bits of the generic shared address),
// vaddr = 64-bit global address, saddr = off (GV mode).
DEV void async_g2l_b128(void* lds_dst, const void* gsrc) {
  unsigned ldsoff = (unsigned)(unsigned long long)lds_dst;
  asm volatile("global_load_async_to_lds_b128 %0, %1, off"
               :: "v"(ldsoff), "v"(gsrc) : "memory");
}

DEV void async_wait_all() {
  asm volatile("s_wait_asynccnt 0x0" ::: "memory");
}

// A-fragment (and, by layout symmetry, B-fragment from row-major [N][K]):
// lane l<16 -> row r0+l, K = {k0..k0+7, k0+16..k0+23}
// lane l>=16 -> row r0+l-16, K = {k0+8..k0+15, k0+24..k0+31}
// Each half is 8 contiguous bf16 = one 16-byte load.
DEV v16bf load_frag(const bf16* base, int ld, int r0, int k0, int lane) {
  int l = lane & 15, h = lane >> 4;
  const bf16* p = base + (size_t)(r0 + l) * ld + (k0 + h * 8);
  v8bf lo = *(const v8bf*)p;
  v8bf hi = *(const v8bf*)(p + 16);
  v16bf a;
#pragma unroll
  for (int i = 0; i < 8; ++i) { a[i] = lo[i]; a[8 + i] = hi[i]; }
  return a;
}

// ---------------------------------------------------------------------------
// Generic WMMA GEMM core: block = 256 threads = 8 waves arranged 4(M) x 2(N).
// Wave tile = 32(M) x 64(N); block tile = 128 x 128.
// A: row-major [M][K] bf16 (lda), W: row-major [N][K] bf16 (ldw) (i.e. math B^T).
// ---------------------------------------------------------------------------
DEV void gemm_core(const bf16* __restrict__ A, int lda,
                   const bf16* __restrict__ W, int ldw,
                   int K, int m0, int n0, int lane, v8f (&acc)[2][4]) {
#pragma unroll
  for (int i = 0; i < 2; ++i)
#pragma unroll
    for (int j = 0; j < 4; ++j) acc[i][j] = vzero();
  for (int k = 0; k < K; k += 32) {
    // speculative prefetch of the next A k-slice (global_prefetch_b8)
    __builtin_prefetch(A + (size_t)(m0 + (lane & 15)) * lda + k + 64, 0, 1);
    v16bf a0 = load_frag(A, lda, m0, k, lane);
    v16bf a1 = load_frag(A, lda, m0 + 16, k, lane);
#pragma unroll
    for (int j = 0; j < 4; ++j) {
      v16bf b = load_frag(W, ldw, n0 + 16 * j, k, lane);
      acc[0][j] = wmma_bf16(a0, b, acc[0][j]);
      acc[1][j] = wmma_bf16(a1, b, acc[1][j]);
    }
  }
}

// ---------------------------------------------------------------------------
// Prep kernels
// ---------------------------------------------------------------------------
__global__ void k_prep_x(const float* __restrict__ x, bf16* __restrict__ xbf) {
  size_t i = ((size_t)blockIdx.x * 256 + threadIdx.x) * 4;
#pragma unroll
  for (int k = 0; k < 4; ++k) xbf[i + k] = (bf16)x[i + k];
}

__global__ void k_prep_w(const float* __restrict__ logA,
                         const float* __restrict__ Bw, const float* __restrict__ Cw,
                         const float* __restrict__ Gw, const float* __restrict__ Ow,
                         bf16* __restrict__ P0, bf16* __restrict__ PT0,
                         bf16* __restrict__ wB, bf16* __restrict__ wC,
                         bf16* __restrict__ wG, bf16* __restrict__ wO) {
  int idx = blockIdx.x * 256 + threadIdx.x;     // < Dm*Dm
  int n = idx >> 9, k = idx & 511;
  float a = __expf(logA[idx]);
  P0[idx]            = (bf16)a;                 // P_1 = A, row-major
  PT0[k * Dm + n]    = (bf16)a;                 // P_1^T
  wB[idx] = (bf16)Bw[idx];
  wC[idx] = (bf16)Cw[idx];
  wG[idx] = (bf16)Gw[idx];
  wO[idx]           = (bf16)Ow[idx];            // out_w is [D][2D]
  wO[idx + Dm * Dm] = (bf16)Ow[idx + Dm * Dm];
}

// ---------------------------------------------------------------------------
// Matrix power doubling: P_{j+m} = P_j @ P_m, j = 1..m (blockIdx.z = j-1).
// Stores result both row-major and transposed (transposed store is 8
// contiguous bf16 per lane = one 16B store).
// ---------------------------------------------------------------------------
__global__ void k_pow(bf16* __restrict__ Parr, bf16* __restrict__ PT, int m) {
  int tid = threadIdx.x, lane = tid & 31, wid = tid >> 5;
  int j = blockIdx.z + 1;
  int m0 = blockIdx.x * 128 + (wid >> 1) * 32;
  int n0 = blockIdx.y * 128 + (wid & 1) * 64;
  const bf16* A = Parr + (size_t)(j - 1) * Dm * Dm;   // P_j rows
  const bf16* W = PT   + (size_t)(m - 1) * Dm * Dm;   // (P_m)^T as [N][K]
  v8f acc[2][4];
  gemm_core(A, Dm, W, Dm, Dm, m0, n0, lane, acc);
  bf16* Po  = Parr + (size_t)(j + m - 1) * Dm * Dm;
  bf16* PTo = PT   + (size_t)(j + m - 1) * Dm * Dm;
  int l = lane & 15, hi = lane >> 4;
#pragma unroll
  for (int i = 0; i < 2; ++i)
#pragma unroll
    for (int f = 0; f < 4; ++f) {
      int n = n0 + 16 * f + l;
      int mb = m0 + 16 * i + hi * 8;
      v8bf pv;
#pragma unroll
      for (int r = 0; r < 8; ++r) {
        bf16 v = (bf16)acc[i][f][r];
        Po[(size_t)(mb + r) * Dm + n] = v;
        pv[r] = v;
      }
      *(v8bf*)&PTo[(size_t)n * Dm + mb] = pv;   // transposed, contiguous
    }
}

// ---------------------------------------------------------------------------
// Bx = x @ B_w^T + B_b   (f32 out)
// ---------------------------------------------------------------------------
__global__ void k_gemm_Bx(const bf16* __restrict__ xbf, const bf16* __restrict__ wB,
                          const float* __restrict__ Bb, float* __restrict__ Bx) {
  int tid = threadIdx.x, lane = tid & 31, wid = tid >> 5;
  int m0 = blockIdx.x * 128 + (wid >> 1) * 32;
  int n0 = blockIdx.y * 128 + (wid & 1) * 64;
  v8f acc[2][4];
  gemm_core(xbf, Dm, wB, Dm, Dm, m0, n0, lane, acc);
  int l = lane & 15, hi = lane >> 4;
#pragma unroll
  for (int i = 0; i < 2; ++i)
#pragma unroll
    for (int f = 0; f < 4; ++f) {
      int n = n0 + 16 * f + l;
      int mb = m0 + 16 * i + hi * 8;
      float bias = Bb[n];
#pragma unroll
      for (int r = 0; r < 8; ++r)
        Bx[(size_t)(mb + r) * Dm + n] = acc[i][f][r] + bias;
    }
}

// ---------------------------------------------------------------------------
// Grouped causal conv (pad K-1 left, truncate): out[t,oc] =
//   sum_{k,ic} w[oc,ic,k] * x[t+k-3, g*64+ic].
// Tile staging uses gfx1250 async global->LDS b128 copies (ASYNCcnt).
// ---------------------------------------------------------------------------
__global__ void k_conv(const float* __restrict__ x, const float* __restrict__ cw,
                       const float* __restrict__ cb, bf16* __restrict__ convb) {
  __shared__ float xs[(64 + 3) * 64];   // [t+3][ic]
  __shared__ float wsh[64 * 64 * 4];    // [oc][ic][k]
  int g = blockIdx.y;
  int m0 = blockIdx.x * 64;             // 64 timesteps, never crosses batch
  int b = m0 / Lseq, t0 = m0 % Lseq;
  int tid = threadIdx.x;
  // weights: 64 KB contiguous -> async b128 per 4 floats
  for (int e = tid * 4; e < 64 * 64 * 4; e += 256 * 4)
    async_g2l_b128(&wsh[e], cw + (size_t)g * 64 * 64 * 4 + e);
  // x tile: rows t0-3 .. t0+63 of this group's 64 channels
  for (int e = tid * 4; e < 67 * 64; e += 256 * 4) {
    int tt = e / 64 - 3 + t0, ic = e & 63;
    if (tt >= 0) {
      async_g2l_b128(&xs[e], &x[((size_t)b * Lseq + tt) * Dm + g * 64 + ic]);
    } else {
      xs[e] = 0.0f; xs[e + 1] = 0.0f; xs[e + 2] = 0.0f; xs[e + 3] = 0.0f;
    }
  }
  async_wait_all();
  __syncthreads();
  int oc = tid & 63, tb = (tid >> 6) * 16;
  float bias = cb[g * 64 + oc];
  for (int ti = 0; ti < 16; ++ti) {
    int tl = tb + ti;
    float s = bias;
#pragma unroll
    for (int k = 0; k < 4; ++k) {
      const float* xr = &xs[(tl + k) * 64];
      const float* wr = &wsh[oc * 256 + k];
#pragma unroll
      for (int ic = 0; ic < 64; ++ic) s += wr[ic * 4] * xr[ic];
    }
    convb[((size_t)(m0 + tl)) * Dm + g * 64 + oc] = (bf16)s;
  }
}

// ---------------------------------------------------------------------------
// comb[:,512:1024] = conv * sigmoid(x @ gate_w^T + g_b)   (bf16 out)
// ---------------------------------------------------------------------------
__global__ void k_gemm_gate(const bf16* __restrict__ xbf, const bf16* __restrict__ wG,
                            const float* __restrict__ gb, const bf16* __restrict__ convb,
                            bf16* __restrict__ comb) {
  int tid = threadIdx.x, lane = tid & 31, wid = tid >> 5;
  int m0 = blockIdx.x * 128 + (wid >> 1) * 32;
  int n0 = blockIdx.y * 128 + (wid & 1) * 64;
  v8f acc[2][4];
  gemm_core(xbf, Dm, wG, Dm, Dm, m0, n0, lane, acc);
  int l = lane & 15, hi = lane >> 4;
#pragma unroll
  for (int i = 0; i < 2; ++i)
#pragma unroll
    for (int f = 0; f < 4; ++f) {
      int n = n0 + 16 * f + l;
      int mb = m0 + 16 * i + hi * 8;
      float bias = gb[n];
#pragma unroll
      for (int r = 0; r < 8; ++r) {
        size_t m = (size_t)(mb + r);
        float sg = 1.0f / (1.0f + __expf(-(acc[i][f][r] + bias)));
        float cv = (float)convb[m * Dm + n];
        comb[m * (2 * Dm) + Dm + n] = (bf16)(cv * sg);
      }
    }
}

// ---------------------------------------------------------------------------
// Scan phase 1: per-chunk local recurrence hloc_j = hloc_{j-1}@A^T + Bx_j.
// One workgroup per chunk; M=16 tile rows = 8 real batches + 8 zero rows.
// h kept in double-buffered LDS; A (bf16, 512KB) streamed from L2 each step.
// ---------------------------------------------------------------------------
__global__ void k_scan1(const float* __restrict__ Bx, const bf16* __restrict__ Abf,
                        bf16* __restrict__ hloc, float* __restrict__ locend) {
  __shared__ bf16 hb[2][16 * Dm];
  int chunk = blockIdx.x;
  int tid = threadIdx.x, lane = tid & 31, wid = tid >> 5;
  int n0 = wid * 64;                       // wave owns 64 state columns
  int l = lane & 15, hi = lane >> 4;
  for (int j = 0; j < CH; ++j) {
    int t = chunk * CH + j;
    v8f acc[4];
#pragma unroll
    for (int f = 0; f < 4; ++f) {
#pragma unroll
      for (int r = 0; r < 8; ++r) {
        float v = 0.0f;
        if (hi == 0)  // rows 0..7 = batches; rows 8..15 stay zero
          v = Bx[((size_t)r * Lseq + t) * Dm + n0 + 16 * f + l];
        acc[f][r] = v;
      }
    }
    if (j > 0) {
      const bf16* hprev = hb[j & 1];
      for (int k = 0; k < Dm; k += 32) {
        v16bf a = load_frag(hprev, Dm, 0, k, lane);
#pragma unroll
        for (int f = 0; f < 4; ++f) {
          v16bf bfrag = load_frag(Abf, Dm, n0 + 16 * f, k, lane);
          acc[f] = wmma_bf16(a, bfrag, acc[f]);
        }
      }
    }
    bf16* hw = hb[(j & 1) ^ 1];
#pragma unroll
    for (int f = 0; f < 4; ++f) {
      int n = n0 + 16 * f + l;
#pragma unroll
      for (int r = 0; r < 8; ++r) {
        int m = hi * 8 + r;
        bf16 hv = (bf16)acc[f][r];
        hw[m * Dm + n] = hv;
        if (hi == 0)
          hloc[((size_t)r * Lseq + t) * Dm + n] = hv;
        if (j == CH - 1)
          locend[((size_t)chunk * 16 + m) * Dm + n] = acc[f][r];
      }
    }
    __syncthreads();
  }
}

// ---------------------------------------------------------------------------
// Scan phase 2: sequential carry E(i) = locend(i) + E(i-1) @ (A^CH)^T.
// Single workgroup, 128 dependent steps.
// ---------------------------------------------------------------------------
__global__ void k_scan2(const float* __restrict__ locend, const bf16* __restrict__ Pc,
                        bf16* __restrict__ Ebf) {
  __shared__ bf16 eb[2][16 * Dm];
  int tid = threadIdx.x, lane = tid & 31, wid = tid >> 5;
  int n0 = wid * 64;
  int l = lane & 15, hi = lane >> 4;
  for (int i = 0; i < NCH; ++i) {
    v8f acc[4];
#pragma unroll
    for (int f = 0; f < 4; ++f) {
#pragma unroll
      for (int r = 0; r < 8; ++r) {
        int m = hi * 8 + r;
        acc[f][r] = locend[((size_t)i * 16 + m) * Dm + n0 + 16 * f + l];
      }
    }
    if (i > 0) {
      const bf16* eprev = eb[i & 1];
      for (int k = 0; k < Dm; k += 32) {
        v16bf a = load_frag(eprev, Dm, 0, k, lane);
#pragma unroll
        for (int f = 0; f < 4; ++f) {
          v16bf bfrag = load_frag(Pc, Dm, n0 + 16 * f, k, lane);
          acc[f] = wmma_bf16(a, bfrag, acc[f]);
        }
      }
    }
    bf16* ew = eb[(i & 1) ^ 1];
#pragma unroll
    for (int f = 0; f < 4; ++f) {
      int n = n0 + 16 * f + l;
#pragma unroll
      for (int r = 0; r < 8; ++r) {
        int m = hi * 8 + r;
        bf16 v = (bf16)acc[f][r];
        ew[m * Dm + n] = v;
        Ebf[((size_t)i * 16 + m) * Dm + n] = v;
      }
    }
    __syncthreads();
  }
}

// ---------------------------------------------------------------------------
// Correction: hs(i,j) = hloc(i,j) + E(i-1) @ (A^{j+1})^T, i>=1.
// One workgroup per (i,j); 8 waves x 64 N-columns; M=16 (8 real rows).
// ---------------------------------------------------------------------------
__global__ void k_corr(bf16* __restrict__ hloc, const bf16* __restrict__ Ebf,
                       const bf16* __restrict__ Parr) {
  int i = blockIdx.x + 1;
  int j = blockIdx.y;
  int tid = threadIdx.x, lane = tid & 31, wid = tid >> 5;
  int n0 = wid * 64;
  const bf16* A = Ebf + (size_t)(i - 1) * 16 * Dm;
  const bf16* W = Parr + (size_t)j * Dm * Dm;     // P_{j+1}
  v8f acc[4];
#pragma unroll
  for (int f = 0; f < 4; ++f) acc[f] = vzero();
  for (int k = 0; k < Dm; k += 32) {
    v16bf a = load_frag(A, Dm, 0, k, lane);
#pragma unroll
    for (int f = 0; f < 4; ++f) {
      v16bf bfrag = load_frag(W, Dm, n0 + 16 * f, k, lane);
      acc[f] = wmma_bf16(a, bfrag, acc[f]);
    }
  }
  int l = lane & 15, hi = lane >> 4;
  if (hi == 0) {   // rows 0..7 = batches
    int t = i * CH + j;
#pragma unroll
    for (int f = 0; f < 4; ++f) {
      int n = n0 + 16 * f + l;
#pragma unroll
      for (int r = 0; r < 8; ++r) {
        size_t idx = ((size_t)r * Lseq + t) * Dm + n;
        hloc[idx] = (bf16)((float)hloc[idx] + acc[f][r]);
      }
    }
  }
}

// ---------------------------------------------------------------------------
// comb[:,0:512] = hs @ C_w^T + C_b + x * D_param   (bf16 out)
// ---------------------------------------------------------------------------
__global__ void k_gemm_C(const bf16* __restrict__ hloc, const bf16* __restrict__ wC,
                         const float* __restrict__ Cb, const float* __restrict__ x,
                         const float* __restrict__ Dp, bf16* __restrict__ comb) {
  int tid = threadIdx.x, lane = tid & 31, wid = tid >> 5;
  int m0 = blockIdx.x * 128 + (wid >> 1) * 32;
  int n0 = blockIdx.y * 128 + (wid & 1) * 64;
  v8f acc[2][4];
  gemm_core(hloc, Dm, wC, Dm, Dm, m0, n0, lane, acc);
  int l = lane & 15, hi = lane >> 4;
#pragma unroll
  for (int i = 0; i < 2; ++i)
#pragma unroll
    for (int f = 0; f < 4; ++f) {
      int n = n0 + 16 * f + l;
      int mb = m0 + 16 * i + hi * 8;
      float bias = Cb[n], dp = Dp[n];
#pragma unroll
      for (int r = 0; r < 8; ++r) {
        size_t m = (size_t)(mb + r);
        float v = acc[i][f][r] + bias + x[m * Dm + n] * dp;
        comb[m * (2 * Dm) + n] = (bf16)v;
      }
    }
}

// ---------------------------------------------------------------------------
// y = comb @ out_w^T + out_b + x   (f32 out; K = 1024)
// ---------------------------------------------------------------------------
__global__ void k_gemm_out(const bf16* __restrict__ comb, const bf16* __restrict__ wO,
                           const float* __restrict__ ob, const float* __restrict__ x,
                           float* __restrict__ y) {
  int tid = threadIdx.x, lane = tid & 31, wid = tid >> 5;
  int m0 = blockIdx.x * 128 + (wid >> 1) * 32;
  int n0 = blockIdx.y * 128 + (wid & 1) * 64;
  v8f acc[2][4];
  gemm_core(comb, 2 * Dm, wO, 2 * Dm, 2 * Dm, m0, n0, lane, acc);
  int l = lane & 15, hi = lane >> 4;
#pragma unroll
  for (int i = 0; i < 2; ++i)
#pragma unroll
    for (int f = 0; f < 4; ++f) {
      int n = n0 + 16 * f + l;
      int mb = m0 + 16 * i + hi * 8;
      float bias = ob[n];
#pragma unroll
      for (int r = 0; r < 8; ++r) {
        size_t m = (size_t)(mb + r);
        y[m * Dm + n] = acc[i][f][r] + bias + x[m * Dm + n];
      }
    }
}

// ---------------------------------------------------------------------------
// Row LayerNorm: one wave per 512-float row.
// ---------------------------------------------------------------------------
__global__ void k_ln(const float* __restrict__ y, const float* __restrict__ lw,
                     const float* __restrict__ lb, float* __restrict__ out) {
  int row = blockIdx.x * 8 + (threadIdx.x >> 5);
  int lane = threadIdx.x & 31;
  const float* yr = y + (size_t)row * Dm;
  float v[16];
#pragma unroll
  for (int i = 0; i < 16; ++i) v[i] = yr[lane * 16 + i];
  float s = 0.0f, ss = 0.0f;
#pragma unroll
  for (int i = 0; i < 16; ++i) { s += v[i]; ss += v[i] * v[i]; }
#pragma unroll
  for (int off = 16; off > 0; off >>= 1) {
    s  += __shfl_xor(s, off, 32);
    ss += __shfl_xor(ss, off, 32);
  }
  float mean = s * (1.0f / Dm);
  float var = ss * (1.0f / Dm) - mean * mean;
  float inv = rsqrtf(var + LN_EPS);
  float* orow = out + (size_t)row * Dm;
#pragma unroll
  for (int i = 0; i < 16; ++i) {
    int n = lane * 16 + i;
    orow[n] = (v[i] - mean) * inv * lw[n] + lb[n];
  }
}

// ---------------------------------------------------------------------------
extern "C" void kernel_launch(void* const* d_in, const int* in_sizes, int n_in,
                              void* d_out, int out_size, void* d_ws, size_t ws_size,
                              hipStream_t stream) {
  const float* x    = (const float*)d_in[0];
  const float* logA = (const float*)d_in[1];
  const float* Bw   = (const float*)d_in[2];
  const float* Bb   = (const float*)d_in[3];
  const float* Cw   = (const float*)d_in[4];
  const float* Cb   = (const float*)d_in[5];
  const float* Dp   = (const float*)d_in[6];
  const float* cw   = (const float*)d_in[7];
  const float* cb   = (const float*)d_in[8];
  const float* Gw   = (const float*)d_in[9];
  const float* gb   = (const float*)d_in[10];
  const float* Ow   = (const float*)d_in[11];
  const float* ob   = (const float*)d_in[12];
  const float* lnw  = (const float*)d_in[13];
  const float* lnb  = (const float*)d_in[14];
  float* out = (float*)d_out;

  // ---- workspace layout (≈522 MiB required) ----
  const size_t MiB = 1ull << 20;
  char* w = (char*)d_ws;
  bf16*  xbf    = (bf16*)(w + 0);            //  64 MiB  x in bf16
  float* BxY    = (float*)(w + 64 * MiB);    // 128 MiB  Bx, later reused as y
  bf16*  hloc   = (bf16*)(w + 192 * MiB);    //  64 MiB  local scan states -> hs
  bf16*  comb   = (bf16*)(w + 256 * MiB);    // 128 MiB  [M,1024] concat (bf16)
  bf16*  convb  = (bf16*)(w + 384 * MiB);    //  64 MiB  conv output (bf16)
  bf16*  Parr   = (bf16*)(w + 448 * MiB);    //  32 MiB  P_1..P_64 row-major
  bf16*  PTarr  = (bf16*)(w + 480 * MiB);    //  32 MiB  P_j^T
  bf16*  wB     = (bf16*)(w + 512 * MiB);
  bf16*  wC     = wB + (size_t)Dm * Dm;
  bf16*  wG     = wC + (size_t)Dm * Dm;
  bf16*  wO     = wG + (size_t)Dm * Dm;      // [D][2D]
  float* locend = (float*)(w + 516 * MiB);   //   4 MiB  chunk-end local states
  bf16*  Ebf    = (bf16*)(w + 520 * MiB);    //   2 MiB  global chunk-end states

  // ---- pipeline ----
  k_prep_x<<<(Mtot * Dm) / 1024, 256, 0, stream>>>(x, xbf);
  k_prep_w<<<(Dm * Dm) / 256, 256, 0, stream>>>(logA, Bw, Cw, Gw, Ow,
                                                Parr, PTarr, wB, wC, wG, wO);
  for (int m = 1; m < CH; m <<= 1)           // P_2..P_64 in 6 doubling rounds
    k_pow<<<dim3(Dm / 128, Dm / 128, m), 256, 0, stream>>>(Parr, PTarr, m);

  k_gemm_Bx<<<dim3(Mtot / 128, Dm / 128), 256, 0, stream>>>(xbf, wB, Bb, BxY);
  k_conv<<<dim3(Mtot / 64, Gc), 256, 0, stream>>>(x, cw, cb, convb);
  k_gemm_gate<<<dim3(Mtot / 128, Dm / 128), 256, 0, stream>>>(xbf, wG, gb, convb, comb);

  k_scan1<<<NCH, 256, 0, stream>>>(BxY, Parr, hloc, locend);
  k_scan2<<<1, 256, 0, stream>>>(locend, Parr + (size_t)(CH - 1) * Dm * Dm, Ebf);
  k_corr<<<dim3(NCH - 1, CH), 256, 0, stream>>>(hloc, Ebf, Parr);

  k_gemm_C<<<dim3(Mtot / 128, Dm / 128), 256, 0, stream>>>(hloc, wC, Cb, x, Dp, comb);
  k_gemm_out<<<dim3(Mtot / 128, Dm / 128), 256, 0, stream>>>(comb, wO, ob, x, BxY);
  k_ln<<<Mtot / 8, 256, 0, stream>>>(BxY, lnw, lnb, out);
}
